// CrossLingualAttention_34024730919223
// MI455X (gfx1250) — matrix operand order repaired and checked by reading, not compiled, not run on hardware
//
#include <hip/hip_runtime.h>

// CrossLingualAttention for MI455X (gfx1250, wave32, WMMA f32_16x16x32_f16).
// All matmuls run on v_wmma_f32_16x16x32_f16 with f32 accumulation; softmax
// and LayerNorm stay in f32. GEMM is double-buffered in LDS with one barrier
// per k-step and 4 WMMAs per wave per k-step (32x32 output tile per wave).

#define B_  8
#define S_  768
#define H_  768
#define NH_ 12
#define HD_ 64
#define NL_ 5

typedef __attribute__((ext_vector_type(16))) _Float16 v16h;
typedef __attribute__((ext_vector_type(8)))  _Float16 v8h;
typedef __attribute__((ext_vector_type(8)))  float    v8f;
typedef __attribute__((ext_vector_type(4)))  float    v4f;

// LDS row strides (halfs). 40 halfs = 80 bytes: keeps every v8h (16B)
// fragment load aligned and staggers rows across the 64 banks.
#define SA 40
#define SB 40

static __device__ __forceinline__ v16h make_v16h(v8h lo, v8h hi) {
  v16h r;
#pragma unroll
  for (int i = 0; i < 8; ++i) { r[i] = lo[i]; r[i + 8] = hi[i]; }
  return r;
}

// A fragment (16x32 f16, ISA 7.12.2 interleave):
//   lanes 0-15: V0-3 k=0..7, V4-7 k=16..23 ; lanes 16-31: +8
static __device__ __forceinline__ v16h fragA(const _Float16* src, int row, int lg) {
  const _Float16* ap = src + row * SA;
  return make_v16h(*(const v8h*)(ap + lg * 8), *(const v8h*)(ap + 16 + lg * 8));
}
// B fragment (32x16 f16, same interleave as A): lanes 0-15 k=0..7,16..23 ; lanes 16-31: +8
static __device__ __forceinline__ v16h fragB(const _Float16* src, int row, int lg) {
  const _Float16* bp = src + row * SB;
  return make_v16h(*(const v8h*)(bp + lg * 8), *(const v8h*)(bp + 16 + lg * 8));
}

// ---------------------------------------------------------------------------
// 128x64 block-tile GEMM core, K in steps of 32, f16 WMMA, f32 accum.
// A:  M x K row-major (batch-resolved pointer)
// B:  TB ? (N x K, computes A @ B^T) : (K x N, computes A @ B)
// C:  M x N row-major; optional bias[N] and residual (same layout as C).
// 256 threads = 8 waves; wave (w&3) -> 32-row slab, (w>>2) -> 32-col slab.
// Double-buffered LDS: next tile's global loads issue before current WMMAs.
// All dims here are multiples of 128/64/32 (768), so no bounds checks.
// ---------------------------------------------------------------------------
template <bool TB>
__device__ __forceinline__ void gemm_tile(
    const float* __restrict__ Ag, const float* __restrict__ Bg,
    const float* __restrict__ biasP, const float* __restrict__ Rg,
    float* __restrict__ Cg, int N, int K, int m0, int n0)
{
  __shared__ _Float16 As[2][128 * SA];
  __shared__ _Float16 Bs[2][64 * SB];

  const int tid  = threadIdx.x;
  const int lane = tid & 31;
  const int lg   = lane >> 4;        // lane group 0/1
  const int ln   = lane & 15;
  const int wave = tid >> 5;
  const int wm   = (wave & 3) << 5;  // 0,32,64,96
  const int wn   = (wave >> 2) << 5; // 0,32

  // staging coordinates (constant per thread)
  const int arow = tid >> 3;              // A row base (step 32)
  const int acol = (tid & 7) << 2;        // A k (float4)
  const int bro  = TB ? (tid >> 3) : (tid >> 4);  // TB: n row ; else k row
  const int bco  = TB ? ((tid & 7) << 2) : ((tid & 15) << 2);

  v4f ra[4], rb[2];

  auto loadTile = [&](int k0) {
    const float* pa = Ag + (size_t)(m0 + arow) * K + k0 + acol;
#pragma unroll
    for (int i = 0; i < 4; ++i) ra[i] = *(const v4f*)(pa + (size_t)(i * 32) * K);
    if (TB) {
      const float* pb = Bg + (size_t)(n0 + bro) * K + k0 + bco;
#pragma unroll
      for (int i = 0; i < 2; ++i) rb[i] = *(const v4f*)(pb + (size_t)(i * 32) * K);
    } else {
      const float* pb = Bg + (size_t)(k0 + bro) * N + n0 + bco;
#pragma unroll
      for (int i = 0; i < 2; ++i) rb[i] = *(const v4f*)(pb + (size_t)(i * 16) * N);
    }
  };

  auto storeTile = [&](int buf) {
#pragma unroll
    for (int i = 0; i < 4; ++i)
#pragma unroll
      for (int j = 0; j < 4; ++j)
        As[buf][(arow + i * 32) * SA + acol + j] = (_Float16)ra[i][j];
    if (TB) {
#pragma unroll
      for (int i = 0; i < 2; ++i)
#pragma unroll
        for (int j = 0; j < 4; ++j)
          Bs[buf][(bro + i * 32) * SB + bco + j] = (_Float16)rb[i][j];
    } else {   // Bs stored n-major: Bs[n][k]
#pragma unroll
      for (int i = 0; i < 2; ++i)
#pragma unroll
        for (int j = 0; j < 4; ++j)
          Bs[buf][(bco + j) * SB + bro + i * 16] = (_Float16)rb[i][j];
    }
  };

  v8f acc00 = {}, acc01 = {}, acc10 = {}, acc11 = {};

  loadTile(0);
  storeTile(0);
  __syncthreads();

  const int KT = K >> 5;
  for (int kt = 0; kt < KT; ++kt) {
    const int cur = kt & 1;
    if (kt + 1 < KT) loadTile((kt + 1) << 5);   // overlap with WMMAs below

    const _Float16* asrc = As[cur];
    const _Float16* bsrc = Bs[cur];
    v16h a0 = fragA(asrc, wm + ln, lg);
    v16h a1 = fragA(asrc, wm + 16 + ln, lg);
    v16h b0 = fragB(bsrc, wn + ln, lg);
    v16h b1 = fragB(bsrc, wn + 16 + ln, lg);

    acc00 = __builtin_amdgcn_wmma_f32_16x16x32_f16(false, a0, false, b0,
                                                   (short)0, acc00, false, false);
    acc01 = __builtin_amdgcn_wmma_f32_16x16x32_f16(false, a0, false, b1,
                                                   (short)0, acc01, false, false);
    acc10 = __builtin_amdgcn_wmma_f32_16x16x32_f16(false, a1, false, b0,
                                                   (short)0, acc10, false, false);
    acc11 = __builtin_amdgcn_wmma_f32_16x16x32_f16(false, a1, false, b1,
                                                   (short)0, acc11, false, false);

    if (kt + 1 < KT) storeTile(cur ^ 1);
    __syncthreads();
  }

  // epilogue: per 16x16 tile, row = v + 8*lg, col = ln
  const int c0 = n0 + wn + ln;
  const int c1 = c0 + 16;
  const float add0 = biasP ? biasP[c0] : 0.0f;
  const float add1 = biasP ? biasP[c1] : 0.0f;
#pragma unroll
  for (int ar = 0; ar < 2; ++ar) {
    const int row = m0 + wm + ar * 16 + (lg << 3);
    const v8f& A0 = ar ? acc10 : acc00;
    const v8f& A1 = ar ? acc11 : acc01;
#pragma unroll
    for (int v = 0; v < 8; ++v) {
      size_t i0 = (size_t)(row + v) * N + c0;
      size_t i1 = (size_t)(row + v) * N + c1;
      float r0 = A0[v] + add0;
      float r1 = A1[v] + add1;
      if (Rg) { r0 += Rg[i0]; r1 += Rg[i1]; }
      Cg[i0] = r0;
      Cg[i1] = r1;
    }
  }
}

// Batched GEMM wrapper. Per-batch B/bias selection: sel = bsel ? bsel[b] : b.
template <bool TB>
__global__ void __launch_bounds__(256) gemm_kernel(
    const float* __restrict__ A, const float* __restrict__ Bm,
    const float* __restrict__ bias, const float* __restrict__ Res,
    float* __restrict__ C, int N, int K,
    long sA, long sB, long biasStride, long sC,
    const int* __restrict__ bsel)
{
  const int bz = blockIdx.z;
  const long sel = bsel ? (long)bsel[bz] : (long)bz;
  const float* Ag    = A + (size_t)bz * sA;
  const float* Bg    = Bm + (size_t)sel * sB;
  const float* biasP = bias ? bias + (size_t)sel * biasStride : nullptr;
  const float* Rg    = Res ? Res + (size_t)bz * sC : nullptr;
  float*       Cg    = C + (size_t)bz * sC;
  gemm_tile<TB>(Ag, Bg, biasP, Rg, Cg, N, K, blockIdx.y * 128, blockIdx.x * 64);
}

// ---------------------------------------------------------------------------
// M-chain: M[b] = Smat[b,0] @ Smat[b,1] @ ... @ Smat[b,B-1]
// Smat[b,j] = I if lang[b]==lang[j], else align[lang[b], lang[j]]
// ---------------------------------------------------------------------------
__global__ void __launch_bounds__(256) m_init(
    const int* __restrict__ lang, const float* __restrict__ algn,
    float* __restrict__ Mout)
{
  const int b  = blockIdx.z;
  const int la = lang[b], l0 = lang[0];
  const int m0 = blockIdx.y * 128, n0 = blockIdx.x * 64;
  float* Mo = Mout + (size_t)b * H_ * H_;
  if (la == l0) {
#pragma unroll
    for (int i = 0; i < 32; ++i) {
      int e = threadIdx.x + (i << 8);
      int r = m0 + (e >> 6), c = n0 + (e & 63);
      Mo[(size_t)r * H_ + c] = (r == c) ? 1.0f : 0.0f;
    }
  } else {
    const float* Sm = algn + ((size_t)la * NL_ + l0) * (size_t)H_ * H_;
#pragma unroll
    for (int i = 0; i < 32; ++i) {
      int e = threadIdx.x + (i << 8);
      int r = m0 + (e >> 6), c = n0 + (e & 63);
      Mo[(size_t)r * H_ + c] = Sm[(size_t)r * H_ + c];
    }
  }
}

__global__ void __launch_bounds__(256) m_step(
    const int* __restrict__ lang, const float* __restrict__ algn,
    const float* __restrict__ Min, float* __restrict__ Mout, int j)
{
  const int b  = blockIdx.z;
  const int la = lang[b], lj = lang[j];            // block-uniform
  const int m0 = blockIdx.y * 128, n0 = blockIdx.x * 64;
  const float* Mi = Min + (size_t)b * H_ * H_;
  float*       Mo = Mout + (size_t)b * H_ * H_;
  if (la == lj) {                                  // Smat = I -> copy
#pragma unroll
    for (int i = 0; i < 32; ++i) {
      int e = threadIdx.x + (i << 8);
      int r = m0 + (e >> 6), c = n0 + (e & 63);
      Mo[(size_t)r * H_ + c] = Mi[(size_t)r * H_ + c];
    }
  } else {
    const float* Sm = algn + ((size_t)la * NL_ + lj) * (size_t)H_ * H_;
    gemm_tile<false>(Mi, Sm, nullptr, nullptr, Mo, H_, H_, m0, n0);
  }
}

// ---------------------------------------------------------------------------
// Attention: one block per (b, head, 16-query tile).
// Scores -> f32 LDS (16 x 768), f32 softmax, then P @ V with WMMA.
// ---------------------------------------------------------------------------
#define SS 776   // padded score row stride (f32)

__global__ void __launch_bounds__(256) attn_kernel(
    const float* __restrict__ Q, const float* __restrict__ Km,
    const float* __restrict__ V, float* __restrict__ ctx)
{
  const int q0   = blockIdx.x * 16;
  const int h    = blockIdx.y;
  const int b    = blockIdx.z;
  const int tid  = threadIdx.x;
  const int lane = tid & 31;
  const int lg   = lane >> 4;
  const int ln   = lane & 15;
  const int wave = tid >> 5;

  __shared__ _Float16 Qs[16 * 72];     // Q tile, row stride 72 (16B aligned)
  __shared__ float    Sc[16 * SS];     // scores/probabilities
  __shared__ float    red[16 * 16];    // softmax reductions

  // stage Q (16 x 64) f32 -> f16
#pragma unroll
  for (int i = 0; i < 4; ++i) {
    int e = tid + (i << 8);
    int r = e >> 6, c = e & 63;
    Qs[r * 72 + c] = (_Float16)Q[(size_t)(b * S_ + q0 + r) * H_ + h * HD_ + c];
  }
  __syncthreads();

  // A fragments for the whole score row (d = 0..31 and 32..63)
  const _Float16* ap = &Qs[ln * 72];
  v16h a0 = make_v16h(*(const v8h*)(ap + lg * 16), *(const v8h*)(ap + 8 + lg * 16));
  v16h a1 = make_v16h(*(const v8h*)(ap + 32 + lg * 16), *(const v8h*)(ap + 40 + lg * 16));

  // scores: wave w covers keys [w*96, w*96+96)
  for (int t = 0; t < 6; ++t) {
    const int key0 = wave * 96 + t * 16;
    const float* kp = Km + (size_t)(b * S_ + key0 + ln) * H_ + h * HD_;
    v16h bk0, bk1;   // B[d, key] = K[key, d]; lane -> key, elements -> d
#pragma unroll
    for (int qd = 0; qd < 4; ++qd) {
      v4f t0 = *(const v4f*)(kp + lg * 16 + qd * 4);
      v4f t1 = *(const v4f*)(kp + 32 + lg * 16 + qd * 4);
#pragma unroll
      for (int jj = 0; jj < 4; ++jj) {
        bk0[qd * 4 + jj] = (_Float16)t0[jj];
        bk1[qd * 4 + jj] = (_Float16)t1[jj];
      }
    }
    v8f acc = {};
    acc = __builtin_amdgcn_wmma_f32_16x16x32_f16(false, a0, false, bk0,
                                                 (short)0, acc, false, false);
    acc = __builtin_amdgcn_wmma_f32_16x16x32_f16(false, a1, false, bk1,
                                                 (short)0, acc, false, false);
#pragma unroll
    for (int v = 0; v < 8; ++v)
      Sc[(v + 8 * lg) * SS + key0 + ln] = acc[v] * 0.125f;  // 1/sqrt(64)
  }
  __syncthreads();

  // softmax: 16 threads per query row
  {
    const int r = tid >> 4, c = tid & 15;
    float m = -3.0e38f;
    for (int i = c; i < S_; i += 16) m = fmaxf(m, Sc[r * SS + i]);
    red[r * 16 + c] = m;
    __syncthreads();
    float mm = -3.0e38f;
#pragma unroll
    for (int i = 0; i < 16; ++i) mm = fmaxf(mm, red[r * 16 + i]);
    __syncthreads();
    float s = 0.0f;
    for (int i = c; i < S_; i += 16) {
      float e = __expf(Sc[r * SS + i] - mm);
      Sc[r * SS + i] = e;
      s += e;
    }
    red[r * 16 + c] = s;
    __syncthreads();
    float ss = 0.0f;
#pragma unroll
    for (int i = 0; i < 16; ++i) ss += red[r * 16 + i];
    const float inv = 1.0f / ss;
    for (int i = c; i < S_; i += 16) Sc[r * SS + i] *= inv;
  }
  __syncthreads();

  // ctx = P @ V : waves 0..3 each own 16 of the 64 head dims, K = 768.
  if (wave < 4) {
    const int d0 = wave * 16;
    v8f acc = {};
    for (int kt = 0; kt < 24; ++kt) {
      const int key0 = kt * 32;
      // A fragment from f32 probabilities (contiguous k = 16*lg + e, matches B)
      v16h p;
      const float* sp = &Sc[ln * SS + key0];
#pragma unroll
      for (int e = 0; e < 16; ++e) {
        int k = lg * 16 + e;
        p[e] = (_Float16)sp[k];
      }
      // B fragment: B[key, d] = V[key, d]; lane -> d, elements -> key
      v16h bv;
#pragma unroll
      for (int e = 0; e < 16; ++e) {
        int k = key0 + lg * 16 + e;
        bv[e] = (_Float16)V[(size_t)(b * S_ + k) * H_ + h * HD_ + d0 + ln];
      }
      acc = __builtin_amdgcn_wmma_f32_16x16x32_f16(false, p, false, bv,
                                                   (short)0, acc, false, false);
    }
#pragma unroll
    for (int v = 0; v < 8; ++v)
      ctx[(size_t)(b * S_ + q0 + v + 8 * lg) * H_ + h * HD_ + d0 + ln] = acc[v];
  }
}

// ---------------------------------------------------------------------------
// LayerNorm over H=768 (one block per token; input already has residual).
// ---------------------------------------------------------------------------
__global__ void __launch_bounds__(256) ln_kernel(
    const float* __restrict__ X, const float* __restrict__ g,
    const float* __restrict__ be, float* __restrict__ out)
{
  const int row = blockIdx.x;
  const float* x = X + (size_t)row * H_;
  __shared__ float s1[256];
  __shared__ float s2[256];
  float a = 0.f, q = 0.f;
#pragma unroll
  for (int i = 0; i < 3; ++i) {
    float v = x[threadIdx.x + (i << 8)];
    a += v; q += v * v;
  }
  s1[threadIdx.x] = a;
  s2[threadIdx.x] = q;
  __syncthreads();
  for (int off = 128; off > 0; off >>= 1) {
    if (threadIdx.x < off) {
      s1[threadIdx.x] += s1[threadIdx.x + off];
      s2[threadIdx.x] += s2[threadIdx.x + off];
    }
    __syncthreads();
  }
  const float mu  = s1[0] * (1.0f / H_);
  const float var = s2[0] * (1.0f / H_) - mu * mu;
  const float inv = rsqrtf(var + 1e-5f);
#pragma unroll
  for (int i = 0; i < 3; ++i) {
    int c = threadIdx.x + (i << 8);
    out[(size_t)row * H_ + c] = (x[c] - mu) * inv * g[c] + be[c];
  }
}

// ---------------------------------------------------------------------------
extern "C" void kernel_launch(void* const* d_in, const int* in_sizes, int n_in,
                              void* d_out, int out_size, void* d_ws, size_t ws_size,
                              hipStream_t stream)
{
  const float* hs   = (const float*)d_in[0];
  const int*   lang = (const int*)d_in[1];
  /* d_in[2] attention_mask: all ones, unused by reference math */
  const float* Wq   = (const float*)d_in[3];
  const float* bq   = (const float*)d_in[4];
  const float* Wk   = (const float*)d_in[5];
  const float* bk   = (const float*)d_in[6];
  const float* ipw  = (const float*)d_in[7];
  const float* ipb  = (const float*)d_in[8];
  const float* opw  = (const float*)d_in[9];
  const float* opb  = (const float*)d_in[10];
  const float* algn = (const float*)d_in[11];
  const float* pw   = (const float*)d_in[12];
  const float* pb   = (const float*)d_in[13];
  const float* lng  = (const float*)d_in[14];
  const float* lnb  = (const float*)d_in[15];
  float* out = (float*)d_out;

  const size_t NBH = (size_t)B_ * S_ * H_;   // == B*H*H too (S_==H_)
  float* W0 = (float*)d_ws;
  float* W1 = W0 + NBH;
  float* W2 = W1 + NBH;
  float* W3 = W2 + NBH;
  float* W4 = W3 + NBH;

  const long sSH = (long)S_ * H_;
  const long sHH = (long)H_ * H_;
  dim3 blk(256, 1, 1);
  dim3 gG(H_ / 64, S_ / 128, B_);   // (12,6,8) — also valid for H x H chain

  // 1-2: language-selected projections  q,k = hs @ W[lang]^T + b[lang]
  gemm_kernel<true><<<gG, blk, 0, stream>>>(hs, Wq, bq, nullptr, W0, H_, H_,
                                            sSH, sHH, (long)H_, sSH, lang);
  gemm_kernel<true><<<gG, blk, 0, stream>>>(hs, Wk, bk, nullptr, W1, H_, H_,
                                            sSH, sHH, (long)H_, sSH, lang);
  // 3-5: Q, K, V projections
  gemm_kernel<true><<<gG, blk, 0, stream>>>(W0, ipw,           ipb,          nullptr,
                                            W2, H_, H_, sSH, 0L, 0L, sSH, nullptr);
  gemm_kernel<true><<<gG, blk, 0, stream>>>(W1, ipw + sHH,     ipb + H_,     nullptr,
                                            W3, H_, H_, sSH, 0L, 0L, sSH, nullptr);
  gemm_kernel<true><<<gG, blk, 0, stream>>>(hs, ipw + 2 * sHH, ipb + 2 * H_, nullptr,
                                            W4, H_, H_, sSH, 0L, 0L, sSH, nullptr);
  // 6: attention -> ctx in W0
  dim3 gA(S_ / 16, NH_, B_);
  attn_kernel<<<gA, blk, 0, stream>>>(W2, W3, W4, W0);
  // 7: aligned = ctx @ out_proj^T + b -> W1
  gemm_kernel<true><<<gG, blk, 0, stream>>>(W0, opw, opb, nullptr, W1, H_, H_,
                                            sSH, 0L, 0L, sSH, nullptr);
  // 8: alignment-matrix chain (ping-pong W2 <-> W3; Q,K are dead now)
  m_init<<<gG, blk, 0, stream>>>(lang, algn, W2);
  float* bufA = W2;
  float* bufB = W3;
  for (int j = 1; j < B_; ++j) {
    m_step<<<gG, blk, 0, stream>>>(lang, algn, bufA, bufB, j);
    float* t = bufA; bufA = bufB; bufB = t;
  }
  // result is in bufA
  // 9: y = aligned @ M[b] -> W4 (V is dead)
  gemm_kernel<false><<<gG, blk, 0, stream>>>(W1, bufA, nullptr, nullptr, W4, H_, H_,
                                             sSH, sHH, 0L, sSH, nullptr);
  // 10: x = y @ proj^T + proj_b + hidden -> W0
  gemm_kernel<true><<<gG, blk, 0, stream>>>(W4, pw, pb, hs, W0, H_, H_,
                                            sSH, 0L, 0L, sSH, nullptr);
  // 11: LayerNorm -> out
  ln_kernel<<<dim3(B_ * S_, 1, 1), blk, 0, stream>>>(W0, lng, lnb, out);
}
